// RNN_stack_29317446763267
// MI455X (gfx1250) — matrix operand order repaired and checked by reading, not compiled, not run on hardware
//
#include <hip/hip_runtime.h>
#include <hip/hip_bf16.h>
#include <math.h>

// ---------------------------------------------------------------------------
// Problem constants (fixed by the reference)
// ---------------------------------------------------------------------------
#define TDIM   2048
#define BDIM   16
#define INSZ   1024
#define EMBD   512
#define HIDD   512
#define OUTD   1024
#define LAYERS 4

typedef __attribute__((ext_vector_type(16))) __bf16 bf16x16;
typedef __attribute__((ext_vector_type(8)))  __bf16 bf16x8;
typedef __attribute__((ext_vector_type(8)))  float  f32x8;

__device__ __forceinline__ f32x8 wmma_bf16(const bf16x16 a, const bf16x16 b, const f32x8 c) {
  // (neg_a, A, neg_b, B, c_mod, C, reuse_a, reuse_b)
  return __builtin_amdgcn_wmma_f32_16x16x32_bf16(false, a, false, b, (short)0, c, false, false);
}

__device__ __forceinline__ f32x8 zero8() {
  f32x8 z = {0.f, 0.f, 0.f, 0.f, 0.f, 0.f, 0.f, 0.f};
  return z;
}

// ---------------------------------------------------------------------------
// Tiled bf16 WMMA GEMM:  C[rowmap(r)][n] = sum_k A[r][k] * B[k][n]
// MODE 0: rowmap = identity
// MODE 1: r = b*T + t  ->  out row = t*B + b   ([B,T,*] -> [T,B,*])
// MODE 2: r = t*B + b  ->  out row = b*T + t   ([T,B,*] -> [B,T,*])
// Block tile 128x128, K-tile 32, 256 threads (8 waves), wave tile 32x64.
// ---------------------------------------------------------------------------
#define BLK_M 128
#define BLK_N 128
#define BLK_K 32

template <int MODE>
__global__ __launch_bounds__(256) void wmma_gemm_kernel(
    const float* __restrict__ A, const float* __restrict__ Bm,
    float* __restrict__ C, int M, int N, int K)
{
  __shared__ __bf16 As[BLK_M * BLK_K];  // [m][k], 64B rows
  __shared__ __bf16 Bs[BLK_N * BLK_K];  // [n][k] (B transposed), 64B rows

  const int tid  = threadIdx.x;
  const int wid  = tid >> 5;
  const int lane = tid & 31;
  const int hl   = lane >> 4;   // lane half (0: lanes 0-15, 1: lanes 16-31)
  const int l16  = lane & 15;
  const int wm   = wid & 3;     // 4 waves along M -> 32 rows each
  const int wn   = wid >> 2;    // 2 waves along N -> 64 cols each
  const int blk_n = blockIdx.x * BLK_N;
  const int blk_m = blockIdx.y * BLK_M;

  f32x8 acc[2][4];
#pragma unroll
  for (int fm = 0; fm < 2; ++fm)
#pragma unroll
    for (int fn = 0; fn < 4; ++fn) acc[fm][fn] = zero8();

  for (int k0 = 0; k0 < K; k0 += BLK_K) {
    // gfx1250 prefetch of next k-tile (global_prefetch_b8)
    if (k0 + BLK_K < K) {
      __builtin_prefetch(&A[(size_t)(blk_m + (tid >> 1)) * K + k0 + BLK_K], 0, 1);
      __builtin_prefetch(&Bm[(size_t)(k0 + BLK_K + (tid >> 3)) * N + blk_n + (tid & 7) * 16], 0, 1);
    }

    // Stage A tile (128x32 fp32 -> bf16), vectorized float4 loads
#pragma unroll
    for (int i = 0; i < 4; ++i) {
      int f   = tid + 256 * i;       // float4 index, 8 per 32-wide row
      int row = f >> 3;
      int kc  = (f & 7) * 4;
      float4 v = *reinterpret_cast<const float4*>(&A[(size_t)(blk_m + row) * K + k0 + kc]);
      __bf16* d = &As[row * BLK_K + kc];
      d[0] = (__bf16)v.x; d[1] = (__bf16)v.y; d[2] = (__bf16)v.z; d[3] = (__bf16)v.w;
    }
    // Stage B tile (32x128 fp32), transposed into [n][k] bf16
#pragma unroll
    for (int i = 0; i < 4; ++i) {
      int f  = tid + 256 * i;        // float4 index, 32 per 128-wide row
      int kk = f >> 5;
      int nc = (f & 31) * 4;
      float4 v = *reinterpret_cast<const float4*>(&Bm[(size_t)(k0 + kk) * N + blk_n + nc]);
      Bs[(nc + 0) * BLK_K + kk] = (__bf16)v.x;
      Bs[(nc + 1) * BLK_K + kk] = (__bf16)v.y;
      Bs[(nc + 2) * BLK_K + kk] = (__bf16)v.z;
      Bs[(nc + 3) * BLK_K + kk] = (__bf16)v.w;
    }
    __syncthreads();

    // A fragments: per ISA, lane half h holds K = h*8+{0..7} and 16+h*8+{0..7}
    bf16x16 afrag[2];
#pragma unroll
    for (int fm = 0; fm < 2; ++fm) {
      int r = wm * 32 + fm * 16 + l16;
      bf16x8 lo = *reinterpret_cast<const bf16x8*>(&As[r * BLK_K + hl * 8]);
      bf16x8 hi = *reinterpret_cast<const bf16x8*>(&As[r * BLK_K + 16 + hl * 8]);
#pragma unroll
      for (int j = 0; j < 8; ++j) { afrag[fm][j] = lo[j]; afrag[fm][j + 8] = hi[j]; }
    }
    // B fragments: lane half h holds K = h*8+{0..7} and 16+h*8+{0..7} of column (lane%16)
    bf16x16 bfrag[4];
#pragma unroll
    for (int fn = 0; fn < 4; ++fn) {
      int c = wn * 64 + fn * 16 + l16;
      bf16x8 lo = *reinterpret_cast<const bf16x8*>(&Bs[c * BLK_K + hl * 8]);
      bf16x8 hi = *reinterpret_cast<const bf16x8*>(&Bs[c * BLK_K + 16 + hl * 8]);
#pragma unroll
      for (int j = 0; j < 8; ++j) { bfrag[fn][j] = lo[j]; bfrag[fn][j + 8] = hi[j]; }
    }

#pragma unroll
    for (int fm = 0; fm < 2; ++fm)
#pragma unroll
      for (int fn = 0; fn < 4; ++fn)
        acc[fm][fn] = wmma_bf16(afrag[fm], bfrag[fn], acc[fm][fn]);
    __syncthreads();
  }

  // Epilogue: C/D layout -> element i is row (hl*8 + i), column (lane%16)
#pragma unroll
  for (int fm = 0; fm < 2; ++fm)
#pragma unroll
    for (int fn = 0; fn < 4; ++fn)
#pragma unroll
      for (int i = 0; i < 8; ++i) {
        int gr = blk_m + wm * 32 + fm * 16 + hl * 8 + i;
        int gc = blk_n + wn * 64 + fn * 16 + l16;
        size_t orow;
        if constexpr (MODE == 0) {
          orow = (size_t)gr;
        } else if constexpr (MODE == 1) {
          int b = gr / TDIM, t = gr % TDIM;
          orow = (size_t)t * BDIM + b;
        } else {
          int t = gr / BDIM, b = gr % BDIM;
          orow = (size_t)b * TDIM + t;
        }
        C[orow * (size_t)N + gc] = acc[fm][fn][i];
      }
}

// ---------------------------------------------------------------------------
// Linear recurrence scan: hs[t] = xw[t] + h_{t-1} @ U     (in place on hs)
// 16 workgroups, WG g owns columns [32g, 32g+32). U slice lives in LDS (bf16),
// fragments preloaded in registers for the entire 2048-step scan.
// Waves: 8 = 2 n-tiles x 4-way K split; deterministic LDS tree reduction.
// Step sync: agent-scope atomic barrier (counter in workspace, memset per launch).
// ---------------------------------------------------------------------------
#define SCAN_WGS 16

__global__ __launch_bounds__(256) void rnn_scan_kernel(
    float* __restrict__ hs,            // [T][B][HID]; contains xw on entry
    const float* __restrict__ Uw,      // [HID][HID]
    const float* __restrict__ h0,      // [HID]
    unsigned* __restrict__ counter)
{
  __shared__ __bf16 Ut[32 * HIDD];     // [n][k] slice of U, 32 KB
  __shared__ float  red[8][256];       // per-wave partials, 8 KB

  const int tid  = threadIdx.x;
  const int wid  = tid >> 5;
  const int lane = tid & 31;
  const int hl   = lane >> 4;
  const int l16  = lane & 15;
  const int ntile = wid >> 2;          // 0..1  -> 16 columns each
  const int kq    = wid & 3;           // 0..3  -> 128 K each
  const int n0    = blockIdx.x * 32;

  // Stage U[:, n0:n0+32] -> Ut[n][k] (bf16), one time
  for (int i = tid; i < 32 * HIDD; i += 256) {
    int k = i >> 5;
    int n = i & 31;
    Ut[n * HIDD + k] = (__bf16)Uw[(size_t)k * HIDD + n0 + n];
  }
  __syncthreads();

  // Preload this wave's 4 B fragments (resident for the whole scan)
  bf16x16 bfrag[4];
#pragma unroll
  for (int kt = 0; kt < 4; ++kt) {
    int c  = ntile * 16 + l16;
    int kb = (kq * 4 + kt) * 32;
    bf16x8 lo = *reinterpret_cast<const bf16x8*>(&Ut[c * HIDD + kb + hl * 8]);
    bf16x8 hi = *reinterpret_cast<const bf16x8*>(&Ut[c * HIDD + kb + 16 + hl * 8]);
#pragma unroll
    for (int j = 0; j < 8; ++j) { bfrag[kt][j] = lo[j]; bfrag[kt][j + 8] = hi[j]; }
  }

  unsigned target = SCAN_WGS;
  for (int t = 0; t < TDIM; ++t) {
    f32x8 acc = zero8();
#pragma unroll
    for (int kt = 0; kt < 4; ++kt) {
      int kb = (kq * 4 + kt) * 32;
      // A fragment: M = batch (lane%16), K from h_{t-1} (fp32 in L2 -> bf16)
      const float* ap = (t == 0)
          ? (h0 + kb)  // broadcast h0 over batch
          : (hs + (size_t)(t - 1) * BDIM * HIDD + (size_t)l16 * HIDD + kb);
      float4 a0 = *reinterpret_cast<const float4*>(ap + hl * 8);
      float4 a1 = *reinterpret_cast<const float4*>(ap + hl * 8 + 4);
      float4 a2 = *reinterpret_cast<const float4*>(ap + 16 + hl * 8);
      float4 a3 = *reinterpret_cast<const float4*>(ap + 16 + hl * 8 + 4);
      float tmp[16] = {a0.x, a0.y, a0.z, a0.w, a1.x, a1.y, a1.z, a1.w,
                       a2.x, a2.y, a2.z, a2.w, a3.x, a3.y, a3.z, a3.w};
      bf16x16 af;
#pragma unroll
      for (int j = 0; j < 16; ++j) af[j] = (__bf16)tmp[j];
      acc = wmma_bf16(af, bfrag[kt], acc);
    }

    // Park partials (C layout: row = hl*8+i, col = lane%16)
#pragma unroll
    for (int i = 0; i < 8; ++i) red[wid][(hl * 8 + i) * 16 + l16] = acc[i];
    __syncthreads();

    // Fixed-order K-split reduction + xw add, in place (xw buffer == hs buffer)
    for (int e = tid; e < 512; e += 256) {
      int nt  = e >> 8;
      int idx = e & 255;
      int m = idx >> 4, n = idx & 15;
      float s = red[nt * 4 + 0][idx] + red[nt * 4 + 1][idx] +
                red[nt * 4 + 2][idx] + red[nt * 4 + 3][idx];
      size_t off = (size_t)t * BDIM * HIDD + (size_t)m * HIDD + (n0 + nt * 16 + n);
      hs[off] += s;
    }

    // Grid barrier across the 16 resident workgroups
    __threadfence();
    __syncthreads();
    if (tid == 0) {
      __hip_atomic_fetch_add(counter, 1u, __ATOMIC_RELEASE, __HIP_MEMORY_SCOPE_AGENT);
      while (__hip_atomic_load(counter, __ATOMIC_ACQUIRE, __HIP_MEMORY_SCOPE_AGENT) < target)
        __builtin_amdgcn_s_sleep(2);
    }
    target += SCAN_WGS;
    __syncthreads();
  }
}

// ---------------------------------------------------------------------------
// Softmax over the TIME axis of logits[B][T][OUT]
// ---------------------------------------------------------------------------
__global__ __launch_bounds__(256) void softmax_stats_kernel(
    const float* __restrict__ logits, float* __restrict__ mx, float* __restrict__ sm)
{
  int col = blockIdx.x * 256 + threadIdx.x;   // b*OUT + o
  int b = col / OUTD, o = col % OUTD;
  const float* p = logits + (size_t)b * TDIM * OUTD + o;
  float m = -INFINITY, s = 0.f;
  for (int t = 0; t < TDIM; ++t) {
    float x = p[(size_t)t * OUTD];
    float mn = fmaxf(m, x);
    s = s * __expf(m - mn) + __expf(x - mn);
    m = mn;
  }
  mx[col] = m;
  sm[col] = s;
}

__global__ __launch_bounds__(256) void softmax_norm_kernel(
    float* __restrict__ logits, const float* __restrict__ mx, const float* __restrict__ sm)
{
  size_t idx = (size_t)blockIdx.x * 256 + threadIdx.x;  // over B*T*OUT
  int o = (int)(idx % OUTD);
  size_t bt = idx / OUTD;
  int b = (int)(bt / TDIM);
  int col = b * OUTD + o;
  logits[idx] = __expf(logits[idx] - mx[col]) / sm[col];
}

// ---------------------------------------------------------------------------
// Orchestration
// ---------------------------------------------------------------------------
extern "C" void kernel_launch(void* const* d_in, const int* in_sizes, int n_in,
                              void* d_out, int out_size, void* d_ws, size_t ws_size,
                              hipStream_t stream) {
  (void)in_sizes; (void)n_in; (void)out_size; (void)ws_size;
  const float* x    = (const float*)d_in[0];   // [B,T,IN]
  const float* h0   = (const float*)d_in[1];   // [L,HID]
  const float* E    = (const float*)d_in[2];   // [IN,EMB]
  const float* W    = (const float*)d_in[3];   // [L,EMB,HID]
  const float* U    = (const float*)d_in[4];   // [L,HID,HID]
  const float* V    = (const float*)d_in[5];   // [L,HID,EMB]
  const float* Wout = (const float*)d_in[6];   // [EMB,OUT]
  float* out = (float*)d_out;                  // [B,T,OUT]

  const int BT = BDIM * TDIM;                  // 32768
  char* ws = (char*)d_ws;
  float*    buf0    = (float*)ws;                                   // [B,T,EMB] 64MB
  float*    buf1    = (float*)(ws + (size_t)BT * EMBD * 4);         // [T,B,HID] 64MB
  float*    mxbuf   = (float*)(ws + 2 * (size_t)BT * EMBD * 4);     // [B*OUT]
  float*    smbuf   = mxbuf + (size_t)BDIM * OUTD;                  // [B*OUT]
  unsigned* counter = (unsigned*)(smbuf + (size_t)BDIM * OUTD);

  dim3 blk(256);

  // 1) Embedding: buf0[b*T+t] = x @ E
  wmma_gemm_kernel<0><<<dim3(EMBD / BLK_N, BT / BLK_M), blk, 0, stream>>>(
      x, E, buf0, BT, EMBD, INSZ);

  for (int l = 0; l < LAYERS; ++l) {
    // 2) xw: buf1[t*B+b] = buf0[b*T+t] @ W[l]   (fused BT->TB transpose)
    wmma_gemm_kernel<1><<<dim3(HIDD / BLK_N, BT / BLK_M), blk, 0, stream>>>(
        buf0, W + (size_t)l * EMBD * HIDD, buf1, BT, HIDD, EMBD);

    // 3) sequential scan in place on buf1 (xw -> hs)
    (void)hipMemsetAsync(counter, 0, sizeof(unsigned), stream);
    rnn_scan_kernel<<<dim3(SCAN_WGS), blk, 0, stream>>>(
        buf1, U + (size_t)l * HIDD * HIDD, h0 + (size_t)l * HIDD, counter);

    // 4) projection: buf0[b*T+t] = buf1[t*B+b] @ V[l]  (fused TB->BT transpose)
    wmma_gemm_kernel<2><<<dim3(EMBD / BLK_N, BT / BLK_M), blk, 0, stream>>>(
        buf1, V + (size_t)l * HIDD * EMBD, buf0, BT, EMBD, HIDD);
  }

  // 5) logits into d_out
  wmma_gemm_kernel<0><<<dim3(OUTD / BLK_N, BT / BLK_M), blk, 0, stream>>>(
      buf0, Wout, out, BT, OUTD, EMBD);

  // 6) softmax over time axis, in place on d_out
  softmax_stats_kernel<<<dim3((BDIM * OUTD) / 256), blk, 0, stream>>>(out, mxbuf, smbuf);
  softmax_norm_kernel<<<dim3((unsigned)(((size_t)BT * OUTD) / 256)), blk, 0, stream>>>(out, mxbuf, smbuf);
}
